// GNN_Model_66520453480861
// MI455X (gfx1250) — matrix operand order repaired and checked by reading, not compiled, not run on hardware
//
#include <hip/hip_runtime.h>
#include <hip/hip_bf16.h>
#include <math.h>

// ---------------------------------------------------------------------------
// GEARS-like GNN forward for MI455X (gfx1250), wave32 + bf16 WMMA.
// Big GEMMs run on v_wmma_f32_16x16x32_bf16; BN/ReLU fused into A-frag loads.
// A-fragment rows are CLAMPED (not predicated) so the A stream is pure b128.
// ---------------------------------------------------------------------------

#define NG_C      32
#define NGENES_C  5000
#define H_C       128
#define NTH_C     9
#define NEDGES_C  100000
#define NPP_C     2
#define RBIG_C    (NG_C * NGENES_C)   // 160000

typedef __attribute__((ext_vector_type(16))) __bf16 v16bf;
typedef __attribute__((ext_vector_type(8)))  float  v8f;

// ------------------------------ helpers ------------------------------------

__device__ __forceinline__ unsigned short f2bf(float f) {
  union { float f; unsigned u; } v; v.f = f;
  unsigned r = v.u + 0x7fffu + ((v.u >> 16) & 1u);
  return (unsigned short)(r >> 16);
}

__device__ __forceinline__ float4 ld4(const float* p) {
  return *reinterpret_cast<const float4*>(p);
}

template <int MODE_>
__device__ __forceinline__ float4 xform4(float4 a, const float* ss, int K, int k) {
  if (MODE_ != 0) {
    float4 sc = ld4(ss + k);
    float4 sh = ld4(ss + K + k);
    a.x = a.x * sc.x + sh.x;  a.y = a.y * sc.y + sh.y;
    a.z = a.z * sc.z + sh.z;  a.w = a.w * sc.w + sh.w;
    if (MODE_ == 1) {
      a.x = fmaxf(a.x, 0.0f); a.y = fmaxf(a.y, 0.0f);
      a.z = fmaxf(a.z, 0.0f); a.w = fmaxf(a.w, 0.0f);
    }
  }
  return a;
}

// Pack row-major f32 weight W[K,N] into the wave32 WMMA B-fragment layout:
// out[(((kt*NT + nt)*32 + lane)*16 + e)] = bf16( W[kt*32 + (e&8)*2 + (lane>>4)*8 + (e&7)][nt*16 + (lane&15)] )
__global__ void pack_w_bf16(const float* __restrict__ W, unsigned short* __restrict__ out,
                            int K, int N) {
  int tid = blockIdx.x * blockDim.x + threadIdx.x;
  int total = K * N;
  if (tid >= total) return;
  int e    = tid & 15;
  int lane = (tid >> 4) & 31;
  int rest = tid >> 9;
  int NT = N >> 4;
  int nt = rest % NT;
  int kt = rest / NT;
  int lo = lane & 15, hi = lane >> 4;
  int k = kt * 32 + (e & 8) * 2 + hi * 8 + (e & 7);
  int n = nt * 16 + lo;
  out[tid] = f2bf(W[(size_t)k * N + n]);
}

// ------------------------------ WMMA GEMM ----------------------------------
// C[R,N] = op(A[R,K]) @ Wpk + bias.   N = NT_*16 (template), K runtime (mult of 32).
// MODE_: 0 = identity, 1 = relu(a*scale+shift), 2 = a*scale+shift.
// Block = 256 threads = 8 waves; each wave owns one 16-row tile (128 rows/block).
// Out-of-range rows are clamped (duplicate loads, stores guarded) so the A
// stream is unpredicated vector loads.
template <int NT_, int MODE_>
__global__ __launch_bounds__(256) void gemm_wmma(
    const float* __restrict__ A, const unsigned short* __restrict__ Wpk,
    const float* __restrict__ bias, float* __restrict__ C,
    int R, int K, const float* __restrict__ ss) {
  const int N = NT_ * 16;
  int wave = threadIdx.x >> 5;
  int lane = threadIdx.x & 31;
  int lo = lane & 15, hi = lane >> 4;
  int tile_m = blockIdx.x * 128 + wave * 16;
  int row = tile_m + lo;
  int rowc = row < R ? row : (R - 1);          // clamp: A row m only feeds D row m
  const float* arow = A + (size_t)rowc * K;
  const v16bf* wp = reinterpret_cast<const v16bf*>(Wpk);
  int KT = K >> 5;

  v8f acc[NT_];
#pragma unroll
  for (int nt = 0; nt < NT_; ++nt) {
    float bv = bias ? bias[nt * 16 + lo] : 0.0f;
#pragma unroll
    for (int r = 0; r < 8; ++r) acc[nt][r] = bv;
  }

  for (int kt = 0; kt < KT; ++kt) {
    // A fragment: lane holds A[row][kt*32 + hi*8 + 0..7] and [.. + 16 + hi*8 + 0..7]
    int k0 = kt * 32 + hi * 8;     // 32-byte aligned
    int k1 = k0 + 16;
    float4 a0 = ld4(arow + k0);
    float4 a1 = ld4(arow + k0 + 4);
    float4 a2 = ld4(arow + k1);
    float4 a3 = ld4(arow + k1 + 4);
    a0 = xform4<MODE_>(a0, ss, K, k0);
    a1 = xform4<MODE_>(a1, ss, K, k0 + 4);
    a2 = xform4<MODE_>(a2, ss, K, k1);
    a3 = xform4<MODE_>(a3, ss, K, k1 + 4);
    v16bf af;
    af[0]  = (__bf16)a0.x; af[1]  = (__bf16)a0.y; af[2]  = (__bf16)a0.z; af[3]  = (__bf16)a0.w;
    af[4]  = (__bf16)a1.x; af[5]  = (__bf16)a1.y; af[6]  = (__bf16)a1.z; af[7]  = (__bf16)a1.w;
    af[8]  = (__bf16)a2.x; af[9]  = (__bf16)a2.y; af[10] = (__bf16)a2.z; af[11] = (__bf16)a2.w;
    af[12] = (__bf16)a3.x; af[13] = (__bf16)a3.y; af[14] = (__bf16)a3.z; af[15] = (__bf16)a3.w;

    const v16bf* wrow = wp + (size_t)kt * NT_ * 32;
    v16bf bfr[NT_];
#pragma unroll
    for (int nt = 0; nt < NT_; ++nt) bfr[nt] = wrow[nt * 32 + lane];
#pragma unroll
    for (int nt = 0; nt < NT_; ++nt) {
      acc[nt] = __builtin_amdgcn_wmma_f32_16x16x32_bf16(
          false, af, false, bfr[nt], (short)0, acc[nt], false, false);
    }
  }

#pragma unroll
  for (int nt = 0; nt < NT_; ++nt) {
#pragma unroll
    for (int r = 0; r < 8; ++r) {
      int orow = tile_m + r + hi * 8;      // D layout: VGPR r -> M = r + 8*hi
      if (orow < R) C[(size_t)orow * N + nt * 16 + lo] = acc[nt][r];
    }
  }
}

// ------------------------- BatchNorm column stats --------------------------

__global__ void col_stats(const float* __restrict__ X, int R, int C, float w,
                          float* __restrict__ asum, float* __restrict__ asq) {
  int nsub = blockDim.x / C;
  int col  = threadIdx.x % C;
  int sub  = threadIdx.x / C;
  int rbeg = blockIdx.x * 256 + sub;
  int rend = blockIdx.x * 256 + 256; if (rend > R) rend = R;
  float s = 0.0f, q = 0.0f;
  for (int r = rbeg; r < rend; r += nsub) {
    float v = X[(size_t)r * C + col];
    s += v; q += v * v;
  }
  atomicAdd(&asum[col], w * s);
  atomicAdd(&asq[col],  w * q);
}

// scale = g*rsqrt(var+eps); shift = be - mean*scale  (ss = [scale[C], shift[C]])
__global__ void bn_finalize(const float* __restrict__ asum, const float* __restrict__ asq,
                            float cnt, const float* __restrict__ g, const float* __restrict__ be,
                            float* __restrict__ ss, int C) {
  int c = threadIdx.x; if (c >= C) return;
  float mean = asum[c] / cnt;
  float var  = asq[c] / cnt - mean * mean;
  float sc   = g[c] * rsqrtf(var + 1e-5f);
  ss[c] = sc;
  ss[C + c] = be[c] - mean * sc;
}

// BN over M rows in one block (M small, e.g. 32)
__global__ void bn_small(const float* __restrict__ X, int M, int N,
                         const float* __restrict__ g, const float* __restrict__ be,
                         float* __restrict__ ss) {
  int c = threadIdx.x; if (c >= N) return;
  float s = 0.0f, q = 0.0f;
  for (int m = 0; m < M; ++m) { float v = X[(size_t)m * N + c]; s += v; q += v * v; }
  float mean = s / M, var = q / M - mean * mean;
  float sc = g[c] * rsqrtf(var + 1e-5f);
  ss[c] = sc; ss[N + c] = be[c] - mean * sc;
}

__global__ void apply_bn(const float* __restrict__ X, const float* __restrict__ ss,
                         float* __restrict__ Y, int total, int C) {
  int i = blockIdx.x * blockDim.x + threadIdx.x;
  if (i >= total) return;
  int c = i % C;
  Y[i] = X[i] * ss[c] + ss[C + c];
}

// ------------------------------ small GEMMs --------------------------------
// (pert/cross branches: M=32 rows, scalar is plenty)
__global__ void sgemm_small(const float* __restrict__ A, const float* __restrict__ W,
                            const float* __restrict__ b, float* __restrict__ Cm,
                            int M, int K, int N, int mode, const float* __restrict__ ss) {
  int idx = blockIdx.x * blockDim.x + threadIdx.x;
  if (idx >= M * N) return;
  int m = idx / N, n = idx % N;
  float acc = b ? b[n] : 0.0f;
  for (int k = 0; k < K; ++k) {
    float a = A[(size_t)m * K + k];
    if (mode) { a = a * ss[k] + ss[K + k]; if (mode == 1) a = fmaxf(a, 0.0f); }
    acc += a * W[(size_t)k * N + n];
  }
  Cm[idx] = acc;
}

// -------------------------- graph / embedding prep -------------------------

__global__ void renorm_rows(const float* __restrict__ X, float* __restrict__ Y, int C) {
  __shared__ float red[128];
  int r = blockIdx.x, c = threadIdx.x;
  float v = X[(size_t)r * C + c];
  red[c] = v * v; __syncthreads();
  for (int s = 64; s > 0; s >>= 1) { if (c < s) red[c] += red[c + s]; __syncthreads(); }
  float n = sqrtf(red[0]);
  float sc = fminf(1.0f, 1.0f / fmaxf(n, 1e-7f));
  Y[(size_t)r * C + c] = v * sc;
}

__global__ void deg_init(float* d, int n) { int i = blockIdx.x * blockDim.x + threadIdx.x; if (i < n) d[i] = 1.0f; }
__global__ void deg_acc(const int* __restrict__ col, const float* __restrict__ ew, float* d, int E) {
  int e = blockIdx.x * blockDim.x + threadIdx.x; if (e < E) atomicAdd(&d[col[e]], ew[e]);
}
__global__ void deg_rsqrt(float* d, int n) { int i = blockIdx.x * blockDim.x + threadIdx.x; if (i < n) d[i] = rsqrtf(d[i]); }

// scatter: AGG[col] += dis[row]*ew*dis[col] * RNP[row]   (wave per edge, 4 cols/lane)
__global__ void edge_agg(const int* __restrict__ rowi, const int* __restrict__ coli,
                         const float* __restrict__ ew, const float* __restrict__ dis,
                         const float* __restrict__ RNP, float* __restrict__ AGG, int E) {
  int gtid = blockIdx.x * blockDim.x + threadIdx.x;
  int e = gtid >> 5, lane = gtid & 31;
  if (e >= E) return;
  int r = rowi[e], c = coli[e];
  float nrm = dis[r] * ew[e] * dis[c];
  const float* src = RNP + (size_t)r * 128;
  float* dst = AGG + (size_t)c * 128;
  for (int k = lane; k < 128; k += 32) atomicAdd(&dst[k], nrm * src[k]);
}

__global__ void selfloop_add(float* __restrict__ AGG, const float* __restrict__ dis,
                             const float* __restrict__ RNP, int total) {
  int i = blockIdx.x * blockDim.x + threadIdx.x;
  if (i >= total) return;
  int n = i >> 7;
  float d = dis[n];
  AGG[i] += d * d * RNP[i];
}

// base = relu(bn(RN_G)); T0 = base + 0.2*POS0; TR = base + 0.2*POSR
__global__ void base_t0tr(const float* __restrict__ RNG, const float* __restrict__ ss,
                          const float* __restrict__ P0, const float* __restrict__ PR,
                          float* __restrict__ T0, float* __restrict__ TR, int total) {
  int i = blockIdx.x * blockDim.x + threadIdx.x;
  if (i >= total) return;
  int c = i & 127;
  float b = fmaxf(RNG[i] * ss[c] + ss[128 + c], 0.0f);
  T0[i] = b + 0.2f * P0[i];
  TR[i] = b + 0.2f * PR[i];
}

// u[row] = Psel[n] + lvlproj[n*10+bin(row)] + g2l_b1   (bin from 9 sorted thresholds)
__global__ void build_u(const float* __restrict__ P0, const float* __restrict__ PR,
                        const float* __restrict__ lvl, const float* __restrict__ b1,
                        const float* __restrict__ x, const float* __restrict__ th,
                        float* __restrict__ U, int Ng) {
  int row = blockIdx.x;
  int c = threadIdx.x;
  int g = row / Ng, n = row - g * Ng;
  __shared__ int sbin;
  if (c == 0) {
    float xv = x[row]; int b = 0;
    for (int t = 0; t < NTH_C; ++t) b += (xv >= th[t * Ng + n]) ? 1 : 0;
    sbin = b;
  }
  __syncthreads();
  const float* P = (g == 0) ? P0 : PR;
  U[(size_t)row * 128 + c] = P[(size_t)n * 128 + c]
                           + lvl[((size_t)n * 10 + sbin) * 128 + c] + b1[c];
}

// h_pre = bn2(v) + pert_lat[g]
__global__ void hpre_kernel(const float* __restrict__ V, const float* __restrict__ ss,
                            const float* __restrict__ PL, float* __restrict__ Hh,
                            int total, int Ng) {
  int i = blockIdx.x * blockDim.x + threadIdx.x;
  if (i >= total) return;
  int row = i >> 7, c = i & 127;
  int g = row / Ng;
  Hh[i] = V[i] * ss[c] + ss[128 + c] + PL[g * 128 + c];
}

// sum_emb[g] = sum_j renorm(pert_emb[pert_idx[g,j]])
__global__ void pert_sum(const float* __restrict__ pe, const int* __restrict__ pidx,
                         float* __restrict__ se) {
  __shared__ float red[128];
  int g = blockIdx.x, c = threadIdx.x;
  float acc = 0.0f;
  for (int j = 0; j < NPP_C; ++j) {
    int id = pidx[g * NPP_C + j];
    float v = pe[(size_t)id * 128 + c];
    red[c] = v * v; __syncthreads();
    for (int s = 64; s > 0; s >>= 1) { if (c < s) red[c] += red[c + s]; __syncthreads(); }
    float n = sqrtf(red[0]);
    __syncthreads();
    float sc = fminf(1.0f, 1.0f / fmaxf(n, 1e-7f));
    acc += v * sc;
  }
  se[g * 128 + c] = acc;
}

// out1[row] = dot(bn(T2[row]), indv_w1[n]) + indv_b1[n]   (one wave per row)
__global__ void indv1_kernel(const float* __restrict__ T2, const float* __restrict__ ss,
                             const float* __restrict__ iw1, const float* __restrict__ ib1,
                             float* __restrict__ out1, int R, int Ng) {
  int gtid = blockIdx.x * blockDim.x + threadIdx.x;
  int row = gtid >> 5, lane = gtid & 31;
  if (row >= R) return;
  int n = row % Ng;
  float acc = 0.0f;
  for (int c = lane; c < 128; c += 32) {
    float v = T2[(size_t)row * 128 + c] * ss[c] + ss[128 + c];
    acc += v * iw1[(size_t)n * 128 + c];
  }
  for (int off = 16; off > 0; off >>= 1) acc += __shfl_down(acc, off, 32);
  if (lane == 0) out1[row] = acc + ib1[n];
}

// cg1[g,c] = b1[c] + sum_n out1[g,n]*w1[n,c]   (K = NGENES)
__global__ void cgs1_kernel(const float* __restrict__ out1, const float* __restrict__ w1,
                            const float* __restrict__ b1, float* __restrict__ cg1, int Ng) {
  int g = blockIdx.x, c = threadIdx.x;
  float acc = b1[c];
  const float* o = out1 + (size_t)g * Ng;
  for (int n = 0; n < Ng; ++n) acc += o[n] * w1[(size_t)n * 128 + c];
  cg1[g * 128 + c] = acc;
}

// out[row] = out1*iw2[n,0] + dot(cg[g], iw2[n,1:129]) + ib2[n] + x[row]
__global__ void final_kernel(const float* __restrict__ out1, const float* __restrict__ CG,
                             const float* __restrict__ iw2, const float* __restrict__ ib2,
                             const float* __restrict__ x, float* __restrict__ out,
                             int R, int Ng) {
  int gtid = blockIdx.x * blockDim.x + threadIdx.x;
  int row = gtid >> 5, lane = gtid & 31;
  if (row >= R) return;
  int g = row / Ng, n = row % Ng;
  float acc = 0.0f;
  for (int c = lane; c < 128; c += 32)
    acc += CG[g * 128 + c] * iw2[(size_t)n * 129 + 1 + c];
  for (int off = 16; off > 0; off >>= 1) acc += __shfl_down(acc, off, 32);
  if (lane == 0)
    out[row] = out1[row] * iw2[(size_t)n * 129] + acc + ib2[n] + x[row];
}

// =============================== host side =================================

extern "C" void kernel_launch(void* const* d_in, const int* in_sizes, int n_in,
                              void* d_out, int out_size, void* d_ws, size_t ws_size,
                              hipStream_t stream) {
  (void)in_sizes; (void)n_in; (void)out_size; (void)ws_size;
  const int Ng = NGENES_C, R = RBIG_C, E = NEDGES_C;

  // ---- inputs (setup_inputs() order; params dict flattened depth-first) ----
  const float* x        = (const float*)d_in[0];
  const float* ew       = (const float*)d_in[1];
  const float* th       = (const float*)d_in[2];
  const float* gene_emb = (const float*)d_in[3];
  const float* emb_pos  = (const float*)d_in[4];
  const float* pert_emb = (const float*)d_in[5];
  const float* expr_tbl = (const float*)d_in[6];
  const float* bn_emb_g = (const float*)d_in[7];
  const float* bn_emb_b = (const float*)d_in[8];
  const float* sgc_w    = (const float*)d_in[9];
  const float* sgc_b    = (const float*)d_in[10];
  const float* et_w1 = (const float*)d_in[11], *et_b1 = (const float*)d_in[12];
  const float* et_g1 = (const float*)d_in[13], *et_be1 = (const float*)d_in[14];
  const float* et_w2 = (const float*)d_in[15], *et_b2 = (const float*)d_in[16];
  const float* et_g2 = (const float*)d_in[17], *et_be2 = (const float*)d_in[18];
  const float* pf_w1 = (const float*)d_in[19], *pf_b1 = (const float*)d_in[20];
  const float* pf_g1 = (const float*)d_in[21], *pf_be1 = (const float*)d_in[22];
  const float* pf_w2 = (const float*)d_in[23], *pf_b2 = (const float*)d_in[24];
  const float* pf_g2 = (const float*)d_in[25], *pf_be2 = (const float*)d_in[26];
  const float* g2_w1 = (const float*)d_in[27], *g2_b1 = (const float*)d_in[28];
  const float* g2_g1 = (const float*)d_in[29], *g2_be1 = (const float*)d_in[30];
  const float* g2_w2 = (const float*)d_in[31], *g2_b2 = (const float*)d_in[32];
  const float* g2_g2 = (const float*)d_in[33], *g2_be2 = (const float*)d_in[34];
  const float* pl_w1 = (const float*)d_in[35], *pl_b1 = (const float*)d_in[36];
  const float* pl_g1 = (const float*)d_in[37], *pl_be1 = (const float*)d_in[38];
  const float* pl_w2 = (const float*)d_in[39], *pl_b2 = (const float*)d_in[40];
  const float* pl_g2 = (const float*)d_in[41], *pl_be2 = (const float*)d_in[42];
  const float* bn_pb_g = (const float*)d_in[43], *bn_pb_b = (const float*)d_in[44];
  const float* rc_w1 = (const float*)d_in[45], *rc_b1 = (const float*)d_in[46];
  const float* rc_g1 = (const float*)d_in[47], *rc_be1 = (const float*)d_in[48];
  const float* rc_w2 = (const float*)d_in[49], *rc_b2 = (const float*)d_in[50];
  const float* rc_g2 = (const float*)d_in[51], *rc_be2 = (const float*)d_in[52];
  const float* iw1   = (const float*)d_in[53], *ib1 = (const float*)d_in[54];
  const float* cg_w1 = (const float*)d_in[55], *cg_b1 = (const float*)d_in[56];
  const float* cg_g1 = (const float*)d_in[57], *cg_be1 = (const float*)d_in[58];
  const float* cg_w2 = (const float*)d_in[59], *cg_b2 = (const float*)d_in[60];
  const float* cg_g2 = (const float*)d_in[61], *cg_be2 = (const float*)d_in[62];
  const float* iw2   = (const float*)d_in[63], *ib2 = (const float*)d_in[64];
  const int* pidx = (const int*)d_in[65];
  const int* gc   = (const int*)d_in[66];

  // ---- workspace carve ----
  char* wsb = (char*)d_ws;
  size_t off = 0;
  auto af = [&](size_t n) -> float* {
    float* p = (float*)(wsb + off); off += ((n * 4 + 255) & ~(size_t)255); return p;
  };
  auto au = [&](size_t n) -> unsigned short* {
    unsigned short* p = (unsigned short*)(wsb + off); off += ((n * 2 + 255) & ~(size_t)255); return p;
  };
  const size_t NH = (size_t)Ng * 128;
  float* RN_G = af(NH);  float* RN_P = af(NH);  float* DIS = af(Ng);
  float* AGG0 = af(NH);  float* POS0 = af(NH);  float* POSR = af(NH);
  float* T0 = af(NH);    float* TR = af(NH);
  float* Hh0 = af(NH);   float* HhR = af(NH);
  float* F0 = af(NH);    float* FR = af(NH);
  float* P0 = af(NH);    float* PR = af(NH);
  float* LVL = af((size_t)Ng * 10 * 128);
  float* BUFA = af((size_t)R * 128);
  float* BUFC = af((size_t)R * 256);          // first half doubles as V
  float* OUT1 = af(R);
  float* SUME = af(NG_C * 128);
  float* PT1 = af(NG_C * 128); float* PT2 = af(NG_C * 128);
  float* PL1 = af(NG_C * 128); float* PL2 = af(NG_C * 128); float* PLAT = af(NG_C * 128);
  float* CG1 = af(NG_C * 128); float* CG2 = af(NG_C * 128); float* CGF = af(NG_C * 128);
  float* ASUM = af(256); float* ASQ = af(256);
  float* S_EMB = af(256); float* S_ET1 = af(256); float* S_ET2 = af(256);
  float* S_G2L1 = af(256); float* S_G2L2 = af(256); float* S_PB = af(256);
  float* S_R1 = af(512); float* S_R2 = af(256);
  float* S_PF1 = af(256); float* S_PF2 = af(256);
  float* S_PL1 = af(256); float* S_PL2 = af(256);
  float* S_CG1 = af(256); float* S_CG2 = af(256);
  unsigned short* PK_SGC = au(128 * 128);
  unsigned short* PK_ET1 = au(128 * 128); unsigned short* PK_ET2 = au(128 * 128);
  unsigned short* PK_G2A = au(128 * 128); unsigned short* PK_G2B = au(128 * 128);
  unsigned short* PK_G2W2 = au(128 * 128);
  unsigned short* PK_R1 = au(128 * 256);  unsigned short* PK_R2 = au(256 * 128);

  // ---- launch helpers ----
  auto pack = [&](const float* W, unsigned short* o, int K, int N) {
    pack_w_bf16<<<(K * N + 255) / 256, 256, 0, stream>>>(W, o, K, N);
  };
  auto stats2 = [&](const float* X1, float w1, const float* X2, float w2,
                    int Rr, int C, float cnt, const float* g, const float* be, float* ss) {
    hipMemsetAsync(ASUM, 0, C * sizeof(float), stream);
    hipMemsetAsync(ASQ, 0, C * sizeof(float), stream);
    int blocks = (Rr + 255) / 256;
    col_stats<<<blocks, 256, 0, stream>>>(X1, Rr, C, w1, ASUM, ASQ);
    if (X2) col_stats<<<blocks, 256, 0, stream>>>(X2, Rr, C, w2, ASUM, ASQ);
    bn_finalize<<<1, C, 0, stream>>>(ASUM, ASQ, cnt, g, be, ss, C);
  };
  auto g128 = [&](const float* A, const unsigned short* W, const float* b, float* Cc,
                  int Rr, int K, int mode, const float* ss) {
    int blocks = (Rr + 127) / 128;
    if (mode == 0)      gemm_wmma<8, 0><<<blocks, 256, 0, stream>>>(A, W, b, Cc, Rr, K, ss);
    else if (mode == 1) gemm_wmma<8, 1><<<blocks, 256, 0, stream>>>(A, W, b, Cc, Rr, K, ss);
    else                gemm_wmma<8, 2><<<blocks, 256, 0, stream>>>(A, W, b, Cc, Rr, K, ss);
  };

  // ---- 0) pack weights into WMMA B-fragment layout (bf16) ----
  pack(sgc_w, PK_SGC, 128, 128);
  pack(et_w1, PK_ET1, 128, 128);  pack(et_w2, PK_ET2, 128, 128);
  pack(g2_w1, PK_G2A, 128, 128);                 // rows 0..127 of [512,128]
  pack(g2_w1 + 128 * 128, PK_G2B, 128, 128);     // rows 128..255
  pack(g2_w2, PK_G2W2, 128, 128);
  pack(rc_w1, PK_R1, 128, 256);   pack(rc_w2, PK_R2, 256, 128);

  // ---- 1) renorm embeddings; SGConv aggregation ----
  renorm_rows<<<Ng, 128, 0, stream>>>(gene_emb, RN_G, 128);
  renorm_rows<<<Ng, 128, 0, stream>>>(emb_pos, RN_P, 128);
  deg_init<<<(Ng + 255) / 256, 256, 0, stream>>>(DIS, Ng);
  deg_acc<<<(E + 255) / 256, 256, 0, stream>>>(gc + E, ew, DIS, E);   // col = gc[1]
  deg_rsqrt<<<(Ng + 255) / 256, 256, 0, stream>>>(DIS, Ng);
  hipMemsetAsync(AGG0, 0, NH * sizeof(float), stream);
  edge_agg<<<((size_t)E * 32 + 255) / 256, 256, 0, stream>>>(gc, gc + E, ew, DIS, RN_P, AGG0, E);
  selfloop_add<<<(NH + 255) / 256, 256, 0, stream>>>(AGG0, DIS, RN_P, (int)NH);

  // pos0 = AGG0 @ sgc_w + b (graph 0);  posR = RN_P @ sgc_w + b (graphs 1..31)
  g128(AGG0, PK_SGC, sgc_b, POS0, Ng, 128, 0, nullptr);
  g128(RN_P, PK_SGC, sgc_b, POSR, Ng, 128, 0, nullptr);

  // ---- 2) base = relu(bn(emb)); t = base + 0.2*pos ----
  stats2(RN_G, 1.0f, nullptr, 0.0f, Ng, 128, (float)Ng, bn_emb_g, bn_emb_b, S_EMB);
  base_t0tr<<<(NH + 255) / 256, 256, 0, stream>>>(RN_G, S_EMB, POS0, POSR, T0, TR, (int)NH);

  // ---- 3) emb_trans_v2 MLP on the 2 distinct row-sets; weighted BN stats ----
  g128(T0, PK_ET1, et_b1, Hh0, Ng, 128, 0, nullptr);
  g128(TR, PK_ET1, et_b1, HhR, Ng, 128, 0, nullptr);
  stats2(Hh0, 1.0f, HhR, 31.0f, Ng, 128, (float)R, et_g1, et_be1, S_ET1);
  g128(Hh0, PK_ET2, et_b2, F0, Ng, 128, 1, S_ET1);
  g128(HhR, PK_ET2, et_b2, FR, Ng, 128, 1, S_ET1);
  stats2(F0, 1.0f, FR, 31.0f, Ng, 128, (float)R, et_g2, et_be2, S_ET2);
  // project base through gene_2_latent first linear (base part of the concat)
  g128(F0, PK_G2A, nullptr, P0, Ng, 128, 2, S_ET2);
  g128(FR, PK_G2A, nullptr, PR, Ng, 128, 2, S_ET2);
  // level part: all Ng*10 expression-level rows projected once
  g128(expr_tbl, PK_G2B, nullptr, LVL, Ng * 10, 128, 0, nullptr);

  // ---- 4) gene_2_latent over all G*Ng rows ----
  build_u<<<R, 128, 0, stream>>>(P0, PR, LVL, g2_b1, x, th, BUFA, Ng);
  stats2(BUFA, 1.0f, nullptr, 0.0f, R, 128, (float)R, g2_g1, g2_be1, S_G2L1);
  g128(BUFA, PK_G2W2, g2_b2, BUFC /*V*/, R, 128, 1, S_G2L1);
  stats2(BUFC, 1.0f, nullptr, 0.0f, R, 128, (float)R, g2_g2, g2_be2, S_G2L2);

  // ---- 5) perturbation branch (32 distinct rows; BN stats over broadcast == over 32) ----
  pert_sum<<<NG_C, 128, 0, stream>>>(pert_emb, pidx, SUME);
  sgemm_small<<<(NG_C * 128 + 255) / 256, 256, 0, stream>>>(SUME, pf_w1, pf_b1, PT1, NG_C, 128, 128, 0, nullptr);
  bn_small<<<1, 128, 0, stream>>>(PT1, NG_C, 128, pf_g1, pf_be1, S_PF1);
  sgemm_small<<<(NG_C * 128 + 255) / 256, 256, 0, stream>>>(PT1, pf_w2, pf_b2, PT2, NG_C, 128, 128, 1, S_PF1);
  bn_small<<<1, 128, 0, stream>>>(PT2, NG_C, 128, pf_g2, pf_be2, S_PF2);
  sgemm_small<<<(NG_C * 128 + 255) / 256, 256, 0, stream>>>(PT2, pl_w1, pl_b1, PL1, NG_C, 128, 128, 2, S_PF2);
  bn_small<<<1, 128, 0, stream>>>(PL1, NG_C, 128, pl_g1, pl_be1, S_PL1);
  sgemm_small<<<(NG_C * 128 + 255) / 256, 256, 0, stream>>>(PL1, pl_w2, pl_b2, PL2, NG_C, 128, 128, 1, S_PL1);
  bn_small<<<1, 128, 0, stream>>>(PL2, NG_C, 128, pl_g2, pl_be2, S_PL2);
  apply_bn<<<(NG_C * 128 + 255) / 256, 256, 0, stream>>>(PL2, S_PL2, PLAT, NG_C * 128, 128);

  // ---- 6) h = relu(bn_pb(cell_lat + pert_lat)); recovery MLP ----
  hpre_kernel<<<((size_t)R * 128 + 255) / 256, 256, 0, stream>>>(BUFC, S_G2L2, PLAT, BUFA, R * 128, Ng);
  stats2(BUFA, 1.0f, nullptr, 0.0f, R, 128, (float)R, bn_pb_g, bn_pb_b, S_PB);
  {
    int blocks = (R + 127) / 128;
    gemm_wmma<16, 1><<<blocks, 256, 0, stream>>>(BUFA, PK_R1, rc_b1, BUFC, R, 128, S_PB); // [R,256]
  }
  stats2(BUFC, 1.0f, nullptr, 0.0f, R, 256, (float)R, rc_g1, rc_be1, S_R1);
  g128(BUFC, PK_R2, rc_b2, BUFA, R, 256, 1, S_R1);   // t2 in BUFA (h_pre dead)
  stats2(BUFA, 1.0f, nullptr, 0.0f, R, 128, (float)R, rc_g2, rc_be2, S_R2);

  // ---- 7) per-gene decoder 1 + cross-gene state + decoder 2 ----
  indv1_kernel<<<((size_t)R * 32 + 255) / 256, 256, 0, stream>>>(BUFA, S_R2, iw1, ib1, OUT1, R, Ng);
  cgs1_kernel<<<NG_C, 128, 0, stream>>>(OUT1, cg_w1, cg_b1, CG1, Ng);
  bn_small<<<1, 128, 0, stream>>>(CG1, NG_C, 128, cg_g1, cg_be1, S_CG1);
  sgemm_small<<<(NG_C * 128 + 255) / 256, 256, 0, stream>>>(CG1, cg_w2, cg_b2, CG2, NG_C, 128, 128, 1, S_CG1);
  bn_small<<<1, 128, 0, stream>>>(CG2, NG_C, 128, cg_g2, cg_be2, S_CG2);
  apply_bn<<<(NG_C * 128 + 255) / 256, 256, 0, stream>>>(CG2, S_CG2, CGF, NG_C * 128, 128);

  final_kernel<<<((size_t)R * 32 + 255) / 256, 256, 0, stream>>>(
      OUT1, CGF, iw2, ib2, x, (float*)d_out, R, Ng);
}
